// GCN_58411555225962
// MI455X (gfx1250) — matrix operand order repaired and checked by reading, not compiled, not run on hardware
//
#include <hip/hip_runtime.h>
#include <hip/hip_bf16.h>

#ifndef __has_builtin
#define __has_builtin(x) 0
#endif

typedef float  v2f  __attribute__((ext_vector_type(2)));
typedef float  v8f  __attribute__((ext_vector_type(8)));
typedef _Float16 v16h __attribute__((ext_vector_type(16)));

#if __has_builtin(__builtin_amdgcn_wmma_f32_16x16x4_f32)
#define GCN_WMMA_F32 1
#else
#define GCN_WMMA_F32 0
#endif

// Native no-return f32 atomic add (global_atomic_add_u32/f32 path, STOREcnt).
__device__ inline void atomic_add_f32(float* p, float v) {
  unsafeAtomicAdd(p, v);
}

// ---------------------------------------------------------------------------
// One 16-row tile of  D = X[row_base:row_base+16, 0:32] @ W[0:32, col_base:col_base+16]
// X row-major, stride 32 (K=32 fixed). W row-major, stride == ncols; columns
// >= ncols read as zero (padding for the 32x8 W2). Returns 16x16 f32 C/D
// fragment (8 VGPRs / lane).
// ---------------------------------------------------------------------------
__device__ inline v8f tile_gemm32(const float* __restrict__ X, int row_base,
                                  const float* __restrict__ W, int ncols,
                                  int col_base, int lane) {
  const int m  = lane & 15;
  const int hi = lane >> 4;
  v8f acc = {};
#if GCN_WMMA_F32
  // A 16x4 f32: lane m holds row M=m; VGPR0 = K=(hi*2), VGPR1 = K=(hi*2+1)
  // B 4x16 f32: lane m holds col N=m; VGPR0 = K=(hi*2), VGPR1 = K=(hi*2+1)
  const int koff = hi * 2;
  const int col  = col_base + m;
  #pragma unroll
  for (int kb = 0; kb < 32; kb += 4) {
    v2f a = *(const v2f*)(X + (row_base + m) * 32 + kb + koff);  // 2 consecutive K
    v2f b;
    b.x = (col < ncols) ? W[(kb + koff)     * ncols + col] : 0.0f;
    b.y = (col < ncols) ? W[(kb + koff + 1) * ncols + col] : 0.0f;
    acc = __builtin_amdgcn_wmma_f32_16x16x4_f32(false, a, false, b,
                                                (short)0, acc, false, false);
  }
#else
  // Fallback: codegen-confirmed f16 WMMA (K=32 in one op), f32->f16 convert.
  v16h a, b;
  const int col = col_base + m;
  #pragma unroll
  for (int e = 0; e < 16; ++e) {
    int v = e >> 1;
    int k = (v < 4) ? (hi * 8 + 2 * v + (e & 1))
                    : (16 + hi * 8 + 2 * (v - 4) + (e & 1));
    a[e] = (_Float16)X[(row_base + m) * 32 + k];
    int kB = (e < 8) ? (hi * 8 + e) : (16 + hi * 8 + (e - 8));
    b[e] = (col < ncols) ? (_Float16)W[kB * ncols + col] : (_Float16)0.0f;
  }
  acc = __builtin_amdgcn_wmma_f32_16x16x32_f16(false, a, false, b,
                                               (short)0, acc, false, false);
#endif
  return acc;
}

// ---------------------------------------------------------------------------
// Kernels
// ---------------------------------------------------------------------------

// Zero the accumulator region of the workspace, seed d_out with bias b2.
__global__ void gcn_init(float* ws, long long total, float* out,
                         const float* __restrict__ b2, int out_n) {
  long long i = (long long)blockIdx.x * blockDim.x + threadIdx.x;
  long long stride = (long long)gridDim.x * blockDim.x;
  for (long long j = i; j < total; j += stride) ws[j] = 0.0f;
  if (i < out_n) out[i] = b2[i];
}

// deg_out[src[e]] += 1 ; deg_in[dst[e]] += 1
__global__ void gcn_degree(const int* __restrict__ src, const int* __restrict__ dst,
                           float* deg_out, float* deg_in, int e_total) {
  int e = blockIdx.x * blockDim.x + threadIdx.x;
  if (e >= e_total) return;
  atomic_add_f32(&deg_out[src[e]], 1.0f);
  atomic_add_f32(&deg_in[dst[e]], 1.0f);
}

// in-place: deg -> rsqrt(max(deg, 1))
__global__ void gcn_rsqrt(float* rs_out, float* rs_in, int n) {
  int i = blockIdx.x * blockDim.x + threadIdx.x;
  if (i >= n) return;
  rs_out[i] = rsqrtf(fmaxf(rs_out[i], 1.0f));
  rs_in[i]  = rsqrtf(fmaxf(rs_in[i], 1.0f));
}

// h0 = (X @ W1) * rs_out[:,None]   (H = 32 -> two 16-col WMMA tiles per row tile)
__global__ void gcn_gemm1(const float* __restrict__ X, const float* __restrict__ W1,
                          const float* __restrict__ rs_out, float* __restrict__ h0,
                          int ntiles) {
  int wave = blockIdx.x * (blockDim.x >> 5) + (threadIdx.x >> 5);
  if (wave >= ntiles) return;                 // wave-uniform guard, EXEC stays all-1
  int lane = threadIdx.x & 31;
  int row_base = wave * 16;
  v8f c0 = tile_gemm32(X, row_base, W1, 32, 0, lane);
  v8f c1 = tile_gemm32(X, row_base, W1, 32, 16, lane);
  int m = lane & 15, hi = lane >> 4;
  #pragma unroll
  for (int r = 0; r < 8; ++r) {
    int node = row_base + r + hi * 8;         // D layout: M = r + 8*hi, N = m
    float s = rs_out[node];
    h0[node * 32 + m]      = c0[r] * s;
    h0[node * 32 + 16 + m] = c1[r] * s;
  }
}

// h2 = (h1 @ W2) * rs_out[:,None]   (C = 8 -> one 16-col tile, cols 8..15 padded 0)
__global__ void gcn_gemm2(const float* __restrict__ h1, const float* __restrict__ W2,
                          const float* __restrict__ rs_out, float* __restrict__ h2,
                          int ntiles) {
  int wave = blockIdx.x * (blockDim.x >> 5) + (threadIdx.x >> 5);
  if (wave >= ntiles) return;
  int lane = threadIdx.x & 31;
  int row_base = wave * 16;
  v8f c = tile_gemm32(h1, row_base, W2, 8, 0, lane);
  int m = lane & 15, hi = lane >> 4;
  if (m < 8) {
    #pragma unroll
    for (int r = 0; r < 8; ++r) {
      int node = row_base + r + hi * 8;
      h2[node * 8 + m] = c[r] * rs_out[node];
    }
  }
}

// Scalar tail for rows not covered by 16-row WMMA tiles (not launched when N%16==0).
__global__ void gcn_gemm_tail(const float* __restrict__ X, const float* __restrict__ W,
                              const float* __restrict__ rs, float* __restrict__ out,
                              int row_start, int nrows, int ncols) {
  int t = blockIdx.x * blockDim.x + threadIdx.x;
  if (t >= nrows * ncols) return;
  int r = row_start + t / ncols;
  int c = t % ncols;
  float acc = 0.0f;
  for (int k = 0; k < 32; ++k) acc = fmaf(X[r * 32 + k], W[k * ncols + c], acc);
  out[r * ncols + c] = acc * rs[r];
}

// agg[dst[e], :] += h[src[e], :]  for 32-wide features; 8 threads/edge, float4 each.
__global__ void gcn_scatter32(const float* __restrict__ h, const int* __restrict__ src,
                              const int* __restrict__ dst, float* agg, int e_total) {
  long long tid = (long long)blockIdx.x * blockDim.x + threadIdx.x;
  if (tid >= (long long)e_total * 8) return;
  int e = (int)(tid >> 3);
  int chunk = ((int)tid & 7) * 4;
  int s = src[e], d = dst[e];
  float4 v = *(const float4*)(h + (long long)s * 32 + chunk);
  float* base = agg + (long long)d * 32 + chunk;
  atomic_add_f32(base + 0, v.x);
  atomic_add_f32(base + 1, v.y);
  atomic_add_f32(base + 2, v.z);
  atomic_add_f32(base + 3, v.w);
}

// agg[dst[e], :] += h[src[e], :]  for 8-wide features; 2 threads/edge, float4 each.
__global__ void gcn_scatter8(const float* __restrict__ h, const int* __restrict__ src,
                             const int* __restrict__ dst, float* agg, int e_total) {
  long long tid = (long long)blockIdx.x * blockDim.x + threadIdx.x;
  if (tid >= (long long)e_total * 2) return;
  int e = (int)(tid >> 1);
  int chunk = ((int)tid & 1) * 4;
  int s = src[e], d = dst[e];
  float4 v = *(const float4*)(h + (long long)s * 8 + chunk);
  float* base = agg + (long long)d * 8 + chunk;
  atomic_add_f32(base + 0, v.x);
  atomic_add_f32(base + 1, v.y);
  atomic_add_f32(base + 2, v.z);
  atomic_add_f32(base + 3, v.w);
}

// h1 = relu(agg1 * rs_in[:,None] + b1)  (in place on agg1)
__global__ void gcn_relu_bias(float* agg1, const float* __restrict__ rs_in,
                              const float* __restrict__ b1, long long total) {
  long long i = (long long)blockIdx.x * blockDim.x + threadIdx.x;
  if (i >= total) return;
  int n = (int)(i >> 5);
  int f = (int)(i & 31);
  agg1[i] = fmaxf(fmaf(agg1[i], rs_in[n], b1[f]), 0.0f);
}

// out[c] += mean_n(agg2[n,c] * rs_in[n])   (out pre-seeded with b2)
__global__ void gcn_reduce_mean(const float* __restrict__ agg2,
                                const float* __restrict__ rs_in,
                                float* out, int n) {
  __shared__ float sdata[256 * 9];            // stride 9: avoid LDS bank conflicts
  float acc[8] = {0, 0, 0, 0, 0, 0, 0, 0};
  for (int i = blockIdx.x * blockDim.x + threadIdx.x; i < n;
       i += gridDim.x * blockDim.x) {
    float s = rs_in[i];
    #pragma unroll
    for (int c = 0; c < 8; ++c) acc[c] += agg2[i * 8 + c] * s;
  }
  #pragma unroll
  for (int c = 0; c < 8; ++c) sdata[threadIdx.x * 9 + c] = acc[c];
  __syncthreads();
  for (int stride = blockDim.x >> 1; stride > 0; stride >>= 1) {
    if ((int)threadIdx.x < stride) {
      #pragma unroll
      for (int c = 0; c < 8; ++c)
        sdata[threadIdx.x * 9 + c] += sdata[(threadIdx.x + stride) * 9 + c];
    }
    __syncthreads();
  }
  if (threadIdx.x < 8)
    atomic_add_f32(&out[threadIdx.x], sdata[threadIdx.x] * (1.0f / (float)n));
}

// ---------------------------------------------------------------------------
// Host launch
// ---------------------------------------------------------------------------
extern "C" void kernel_launch(void* const* d_in, const int* in_sizes, int n_in,
                              void* d_out, int out_size, void* d_ws, size_t ws_size,
                              hipStream_t stream) {
  const float* in_feat = (const float*)d_in[0];
  const int*   src     = (const int*)d_in[1];
  const int*   dst     = (const int*)d_in[2];
  const float* W1      = (const float*)d_in[3];
  const float* b1      = (const float*)d_in[4];
  const float* W2      = (const float*)d_in[5];
  const float* b2      = (const float*)d_in[6];
  float* out = (float*)d_out;

  const int n = in_sizes[0] / 32;   // nodes
  const int e = in_sizes[1];        // edges

  // Workspace layout: accumulators first (zeroed), then GEMM outputs (overwritten).
  float* ws     = (float*)d_ws;
  float* rs_out = ws;                       // [n]    deg_out -> rsqrt
  float* rs_in  = ws + n;                   // [n]    deg_in  -> rsqrt
  float* agg1   = ws + 2LL * n;             // [n,32] scatter sum -> h1 (in place)
  float* agg2   = ws + 34LL * n;            // [n,8]  scatter sum
  float* h0     = ws + 42LL * n;            // [n,32] (X@W1)*rs_out
  float* h2     = ws + 74LL * n;            // [n,8]  (h1@W2)*rs_out
  const long long zero_total = 42LL * n;    // rs_out, rs_in, agg1, agg2

  const int ntiles = n / 16;
  const int rem    = n - ntiles * 16;
  const int B = 256;

  gcn_init<<<2048, B, 0, stream>>>(ws, zero_total, out, b2, out_size);

  gcn_degree<<<(e + B - 1) / B, B, 0, stream>>>(src, dst, rs_out, rs_in, e);
  gcn_rsqrt<<<(n + B - 1) / B, B, 0, stream>>>(rs_out, rs_in, n);

  // ---- Layer 1: h0 = (X@W1)*rs_out ; agg1 = scatter ; h1 = relu(agg1*rs_in + b1)
  if (ntiles > 0)
    gcn_gemm1<<<(ntiles + 7) / 8, B, 0, stream>>>(in_feat, W1, rs_out, h0, ntiles);
  if (rem > 0)
    gcn_gemm_tail<<<(rem * 32 + B - 1) / B, B, 0, stream>>>(in_feat, W1, rs_out, h0,
                                                            ntiles * 16, rem, 32);
  {
    long long work = (long long)e * 8;
    gcn_scatter32<<<(unsigned)((work + B - 1) / B), B, 0, stream>>>(h0, src, dst, agg1, e);
  }
  {
    long long work = (long long)n * 32;
    gcn_relu_bias<<<(unsigned)((work + B - 1) / B), B, 0, stream>>>(agg1, rs_in, b1, work);
  }

  // ---- Layer 2: h2 = (h1@W2)*rs_out ; agg2 = scatter ; mean pool + b2
  if (ntiles > 0)
    gcn_gemm2<<<(ntiles + 7) / 8, B, 0, stream>>>(agg1, W2, rs_out, h2, ntiles);
  if (rem > 0)
    gcn_gemm_tail<<<(rem * 8 + B - 1) / B, B, 0, stream>>>(agg1, W2, rs_out, h2,
                                                           ntiles * 16, rem, 8);
  {
    long long work = (long long)e * 2;
    gcn_scatter8<<<(unsigned)((work + B - 1) / B), B, 0, stream>>>(h2, src, dst, agg2, e);
  }
  gcn_reduce_mean<<<400, B, 0, stream>>>(agg2, rs_in, out, n);
}
